// LSTM_6597069767124
// MI455X (gfx1250) — matrix operand order repaired and checked by reading, compiled, not run on hardware
//
#include <hip/hip_runtime.h>

typedef __bf16 bf16;
typedef __attribute__((ext_vector_type(8)))  __bf16 v8bf;
typedef __attribute__((ext_vector_type(16))) __bf16 v16bf;
typedef __attribute__((ext_vector_type(8)))  float  v8f;

// Global (address_space(1)) pointer types so laundered pointers still lower to
// global_load_b128 (not flat_load_b128, which also burns DScnt).
typedef const __attribute__((address_space(1))) bf16*  gcbf;
typedef const __attribute__((address_space(1))) v8bf*  gv8p;

#define B_  64
#define T_  512
#define I_  512
#define H_  1024
#define G4  4096
#define NWG 64

// Defeat LICM: round-trip a pointer through an opaque 64-bit SGPR value, then
// rebuild it as an explicit global-addrspace pointer.
__device__ __forceinline__ gcbf launder_g(const bf16* p) {
    unsigned long long v = (unsigned long long)p;
    asm volatile("" : "+s"(v));
    return (gcbf)v;
}
__device__ __forceinline__ gcbf as_g(const bf16* p) {
    return (gcbf)(unsigned long long)p;
}

// ---------- helpers ----------
__device__ __forceinline__ float sigmoidf_(float x) { return 1.f / (1.f + __expf(-x)); }
__device__ __forceinline__ float tanhf_(float x) {
    x = fminf(15.f, fmaxf(-15.f, x));
    float e = __expf(2.f * x);
    return (e - 1.f) / (e + 1.f);
}
__device__ __forceinline__ v16bf pack16(v8bf lo, v8bf hi) {
    return __builtin_shufflevector(lo, hi, 0,1,2,3,4,5,6,7,8,9,10,11,12,13,14,15);
}

// Accumulate one K-segment into two 16x16 tiles that share the A (M) tile.
// A layout per CDNA5 ISA: lane<16 -> rows, k-halves {0..7,16..23}; lane>=16 -> {8..15,24..31}
// B layout: same k-halves as A: lane<16 -> {0..7,16..23}; lane>=16 -> {8..15,24..31} of column n.
__device__ __forceinline__ void mma_seg(gcbf A, long lda,
                                        gcbf W, long ldw,
                                        int K, int n0a, int n0b,
                                        int m0, int lane, v8f& acc0, v8f& acc1)
{
    const int l15 = lane & 15;
    const int hi  = lane >> 4;
    gcbf ap  = A + (long)(m0 + l15) * lda + hi * 8;
    gcbf bpa = W + (long)(n0a + l15) * ldw + hi * 8;
    gcbf bpb = W + (long)(n0b + l15) * ldw + hi * 8;
    for (int k0 = 0; k0 < K; k0 += 32) {
        v16bf a  = pack16(*(gv8p)(ap  + k0), *(gv8p)(ap  + k0 + 16));
        v16bf b0 = pack16(*(gv8p)(bpa + k0), *(gv8p)(bpa + k0 + 16));
        v16bf b1 = pack16(*(gv8p)(bpb + k0), *(gv8p)(bpb + k0 + 16));
        acc0 = __builtin_amdgcn_wmma_f32_16x16x32_bf16(false, a, false, b0, (short)0, acc0, false, false);
        acc1 = __builtin_amdgcn_wmma_f32_16x16x32_bf16(false, a, false, b1, (short)0, acc1, false, false);
    }
}

// One LSTM layer step for this workgroup's 16 h-columns [j0, j0+16).
// gates = Ain @ Wih^T + Hprev @ Whh^T + bias ; then elementwise cell update.
__device__ __forceinline__ void lstm_phase(
    const bf16* __restrict__ Ain, long ldaIn, int Kin,
    const bf16* __restrict__ Wih,            // (4096 x Kin) row-major
    const bf16* __restrict__ Hprev,          // (64 x 1024)
    const bf16* __restrict__ Whh,            // (4096 x 1024) row-major
    const float* __restrict__ bias,          // (4096) combined b_ih + b_hh
    float* __restrict__ Cst,                 // (64 x 1024) cell state (this layer)
    bf16*  __restrict__ Hnext,               // (64 x 1024) bf16 out
    float* __restrict__ Hf32,                // optional f32 copy of h (layer 1)
    float* lds, int j0)
{
    // Opaque (per-step) global pointers for the weights: keeps the weight
    // streams inside the t-loop (read from L2 each step) instead of being
    // hoisted across all 512 steps and spilled to scratch.
    gcbf Wg = launder_g(Wih);
    gcbf Ug = launder_g(Whh);

    const int tid  = threadIdx.x;
    const int wave = tid >> 5;
    const int lane = tid & 31;
    const int m0   = (wave & 3) << 4;        // M tile base (batch rows)
    const int g0   = wave >> 2;              // gate pair {g0, g0+2}
    const int n0a  = g0 * H_ + j0;
    const int n0b  = (g0 + 2) * H_ + j0;

    v8f acc0 = {}; v8f acc1 = {};
    mma_seg(as_g(Ain),   ldaIn, Wg, (long)Kin, Kin, n0a, n0b, m0, lane, acc0, acc1);
    mma_seg(as_g(Hprev), H_,    Ug, (long)H_,  H_,  n0a, n0b, m0, lane, acc0, acc1);

    // Scatter accumulators to LDS: element (M,N) -> M = m0 + v + 8*(lane>=16), N = lane&15
    const int col   = lane & 15;
    const int rbase = m0 + ((lane >> 4) << 3);
    #pragma unroll
    for (int v = 0; v < 8; ++v) {
        lds[(rbase + v) * 64 + g0 * 16 + col]       = acc0[v];
        lds[(rbase + v) * 64 + (g0 + 2) * 16 + col] = acc1[v];
    }
    __syncthreads();

    // Elementwise cell update: 64 rows x 16 cols, 256 threads -> 4 elems each
    const int cj = tid & 15;
    const int jg = j0 + cj;
    const float bi = bias[0 * H_ + jg];
    const float bf = bias[1 * H_ + jg];
    const float bg = bias[2 * H_ + jg];
    const float bo = bias[3 * H_ + jg];
    for (int r = tid >> 4; r < 64; r += 16) {
        float gi = lds[r * 64 +  0 + cj] + bi;
        float gf = lds[r * 64 + 16 + cj] + bf;
        float gg = lds[r * 64 + 32 + cj] + bg;
        float go = lds[r * 64 + 48 + cj] + bo;
        float i = sigmoidf_(gi);
        float f = sigmoidf_(gf);
        float g = tanhf_(gg);
        float o = sigmoidf_(go);
        long idx = (long)r * H_ + jg;
        float c = f * Cst[idx] + i * g;
        Cst[idx] = c;
        float h = o * tanhf_(c);
        Hnext[idx] = (bf16)h;
        if (Hf32) Hf32[idx] = h;
    }
    __syncthreads();
}

__device__ __forceinline__ void grid_barrier(unsigned* cnt, unsigned* gen) {
    __threadfence();
    __syncthreads();
    if (threadIdx.x == 0) {
        unsigned g = __hip_atomic_load(gen, __ATOMIC_ACQUIRE, __HIP_MEMORY_SCOPE_AGENT);
        unsigned a = __hip_atomic_fetch_add(cnt, 1u, __ATOMIC_ACQ_REL, __HIP_MEMORY_SCOPE_AGENT);
        if (a == NWG - 1) {
            __hip_atomic_store(cnt, 0u, __ATOMIC_RELAXED, __HIP_MEMORY_SCOPE_AGENT);
            __hip_atomic_fetch_add(gen, 1u, __ATOMIC_RELEASE, __HIP_MEMORY_SCOPE_AGENT);
        } else {
            while (__hip_atomic_load(gen, __ATOMIC_ACQUIRE, __HIP_MEMORY_SCOPE_AGENT) == g)
                __builtin_amdgcn_s_sleep(1);
        }
    }
    __syncthreads();
}

// ---------- kernels ----------
__global__ void f32_to_bf16_k(const float* __restrict__ src, bf16* __restrict__ dst, long n) {
    for (long i = (long)blockIdx.x * blockDim.x + threadIdx.x; i < n; i += (long)gridDim.x * blockDim.x)
        dst[i] = (bf16)src[i];
}
__global__ void add_bias_k(const float* __restrict__ a, const float* __restrict__ b,
                           float* __restrict__ o, int n) {
    int i = blockIdx.x * blockDim.x + threadIdx.x;
    if (i < n) o[i] = a[i] + b[i];
}
__global__ void zero_u32_k(unsigned* __restrict__ p, long n) {
    for (long i = (long)blockIdx.x * blockDim.x + threadIdx.x; i < n; i += (long)gridDim.x * blockDim.x)
        p[i] = 0u;
}

__global__ void __launch_bounds__(256, 1) lstm_persistent(
    const bf16* __restrict__ xbf,
    const bf16* __restrict__ Wih0, const bf16* __restrict__ Whh0, const float* __restrict__ b0,
    const bf16* __restrict__ Wih1, const bf16* __restrict__ Whh1, const float* __restrict__ b1,
    bf16* __restrict__ h0buf, bf16* __restrict__ h1buf,
    float* __restrict__ c0, float* __restrict__ c1, float* __restrict__ h1f32,
    unsigned* __restrict__ bar)
{
    __shared__ float lds[64 * 64];
    const int j0 = blockIdx.x * 16;
    unsigned* cnt = bar;
    unsigned* gen = bar + 1;
    const long HB = (long)B_ * H_;   // 65536 elements per h buffer

    for (int t = 0; t < T_; ++t) {
        const bf16* h0_prev = h0buf + (long)(t & 1) * HB;
        bf16*       h0_next = h0buf + (long)((t + 1) & 1) * HB;
        const bf16* h1_prev = h1buf + (long)(t & 1) * HB;
        bf16*       h1_next = h1buf + (long)((t + 1) & 1) * HB;

        // Prefetch next step's x strip for this thread's rows (only stream not
        // guaranteed L2-resident).
        if (t + 1 < T_) {
            const bf16* nx = xbf + (long)(t + 1) * I_ + (long)(threadIdx.x & 63) * (long)T_ * I_;
            __builtin_prefetch(nx, 0, 1);
        }

        // Layer 0: gates = x_t @ Wih0^T + h0 @ Whh0^T + b0
        lstm_phase(xbf + (long)t * I_, (long)T_ * I_, I_,
                   Wih0, h0_prev, Whh0, b0, c0, h0_next, nullptr, lds, j0);
        grid_barrier(cnt, gen);

        // Layer 1: gates = h0_t @ Wih1^T + h1 @ Whh1^T + b1
        lstm_phase(h0_next, (long)H_, H_,
                   Wih1, h1_prev, Whh1, b1, c1, h1_next,
                   (t == T_ - 1) ? h1f32 : nullptr, lds, j0);
        grid_barrier(cnt, gen);
    }
}

__global__ void out_proj_k(const float* __restrict__ h, const float* __restrict__ Wout,
                           const float* __restrict__ bout, float* __restrict__ out) {
    __shared__ float red[256];
    float s = 0.f;
    for (int k = threadIdx.x; k < H_; k += 256)
        s += h[(long)blockIdx.x * H_ + k] * Wout[k];
    red[threadIdx.x] = s;
    __syncthreads();
    for (int off = 128; off > 0; off >>= 1) {
        if (threadIdx.x < off) red[threadIdx.x] += red[threadIdx.x + off];
        __syncthreads();
    }
    if (threadIdx.x == 0) out[blockIdx.x] = sigmoidf_(red[0] + bout[0]);
}

// ---------- launch ----------
extern "C" void kernel_launch(void* const* d_in, const int* in_sizes, int n_in,
                              void* d_out, int out_size, void* d_ws, size_t ws_size,
                              hipStream_t stream) {
    const float* x     = (const float*)d_in[0];
    const float* W_ih0 = (const float*)d_in[1];
    const float* W_hh0 = (const float*)d_in[2];
    const float* b_ih0 = (const float*)d_in[3];
    const float* b_hh0 = (const float*)d_in[4];
    const float* W_ih1 = (const float*)d_in[5];
    const float* W_hh1 = (const float*)d_in[6];
    const float* b_ih1 = (const float*)d_in[7];
    const float* b_hh1 = (const float*)d_in[8];
    const float* W_out = (const float*)d_in[9];
    const float* b_out = (const float*)d_in[10];
    float* out = (float*)d_out;

    char* ws = (char*)d_ws;
    // workspace layout (bytes)
    bf16*  Wih0b = (bf16*)(ws + 0);              // 4096*512*2  = 4,194,304
    bf16*  Whh0b = (bf16*)(ws + 4194304);        // 4096*1024*2 = 8,388,608
    bf16*  Wih1b = (bf16*)(ws + 12582912);       // 8,388,608
    bf16*  Whh1b = (bf16*)(ws + 20971520);       // 8,388,608
    bf16*  xbf   = (bf16*)(ws + 29360128);       // 64*512*512*2 = 33,554,432
    float* b0    = (float*)(ws + 62914560);      // 16,384
    float* b1    = (float*)(ws + 62930944);      // 16,384
    bf16*  h0buf = (bf16*)(ws + 62947328);       // 2*64*1024*2 = 262,144
    bf16*  h1buf = (bf16*)(ws + 63209472);       // 262,144
    float* c0    = (float*)(ws + 63471616);      // 262,144
    float* c1    = (float*)(ws + 63733760);      // 262,144
    float* h1f   = (float*)(ws + 63995904);      // 262,144
    unsigned* bar = (unsigned*)(ws + 64258048);  // 8

    // Convert weights + x to bf16
    f32_to_bf16_k<<<1024, 256, 0, stream>>>(W_ih0, Wih0b, (long)G4 * I_);
    f32_to_bf16_k<<<1024, 256, 0, stream>>>(W_hh0, Whh0b, (long)G4 * H_);
    f32_to_bf16_k<<<1024, 256, 0, stream>>>(W_ih1, Wih1b, (long)G4 * H_);
    f32_to_bf16_k<<<1024, 256, 0, stream>>>(W_hh1, Whh1b, (long)G4 * H_);
    f32_to_bf16_k<<<2048, 256, 0, stream>>>(x,     xbf,   (long)B_ * T_ * I_);

    // Combined biases
    add_bias_k<<<16, 256, 0, stream>>>(b_ih0, b_hh0, b0, G4);
    add_bias_k<<<16, 256, 0, stream>>>(b_ih1, b_hh1, b1, G4);

    // Zero state region: h0buf..bar inclusive (1,310,728 bytes = 327,682 u32)
    zero_u32_k<<<512, 256, 0, stream>>>((unsigned*)(ws + 62947328), 327682);

    // Persistent two-layer LSTM scan
    lstm_persistent<<<NWG, 256, 0, stream>>>(xbf, Wih0b, Whh0b, b0, Wih1b, Whh1b, b1,
                                             h0buf, h1buf, c0, c1, h1f, bar);

    // Output projection + sigmoid: (64 x 1024) @ (1024) -> 64
    out_proj_k<<<B_, 256, 0, stream>>>(h1f, W_out, b_out, out);
    (void)in_sizes; (void)n_in; (void)out_size; (void)ws_size;
}
